// MeshConvNet_42949673552
// MI455X (gfx1250) — matrix operand order repaired and checked by reading, not compiled, not run on hardware
//
#include <hip/hip_runtime.h>
#include <hip/hip_bf16.h>

typedef _Float16 half_t;
typedef _Float16 __attribute__((ext_vector_type(16))) v16h;
typedef _Float16 __attribute__((ext_vector_type(8)))  h8;
typedef float    __attribute__((ext_vector_type(8)))  v8f;

#define NB 16            // batch
#define TILE_E 32        // edges per workgroup

static __device__ inline v16h concat8(h8 a, h8 b) {
    return __builtin_shufflevector(a, b, 0, 1, 2, 3, 4, 5, 6, 7,
                                         8, 9, 10, 11, 12, 13, 14, 15);
}

// ---------------------------------------------------------------------------
// fp32 -> f16 convert (network input)
// ---------------------------------------------------------------------------
__global__ void cvt_f32_f16(const float* __restrict__ in, half_t* __restrict__ out, int n) {
    int i = blockIdx.x * blockDim.x + threadIdx.x;
    if (i < n) out[i] = (half_t)in[i];
}

// ---------------------------------------------------------------------------
// Weight reorder: W (Cout, Cin, 5) fp32  ->  Wk (Cout, Kpad) f16
// K ordering: chunk ci of 32 channels -> k = ci*160 + s*32 + cl  (Cin%32==0)
// Cin==5 special case: Kpad=32, k = s*5 + c, zero padded to 32.
// ---------------------------------------------------------------------------
__global__ void prep_w(const float* __restrict__ W, half_t* __restrict__ Wk,
                       int Cout, int Cin, int Kpad) {
    int i = blockIdx.x * blockDim.x + threadIdx.x;
    if (i >= Cout * Kpad) return;
    int o = i / Kpad, k = i % Kpad;
    float v = 0.f;
    if (Cin == 5) {
        if (k < 25) { int s = k / 5, c = k % 5; v = W[((size_t)o * 5 + c) * 5 + s]; }
    } else {
        int ci = k / 160, r = k % 160, s = r / 32, c = ci * 32 + (r % 32);
        v = W[((size_t)o * Cin + c) * 5 + s];
    }
    Wk[(size_t)o * Kpad + k] = (half_t)v;
}

// ---------------------------------------------------------------------------
// Per-channel BN stats over relu(Y), Y:(B,C,E) f16. Produces fused affine:
//   scsh[c]   = gamma*rsqrt(var+eps)
//   scsh[C+c] = beta - mean*scale
// ---------------------------------------------------------------------------
__global__ void bn_stats(const half_t* __restrict__ Y, const float* __restrict__ gamma,
                         const float* __restrict__ beta, float* __restrict__ scsh,
                         int C, int E) {
    int ch = blockIdx.x, tid = threadIdx.x;
    float s0 = 0.f, s1 = 0.f;
    for (int b = 0; b < NB; ++b) {
        const half_t* p = Y + ((size_t)b * C + ch) * E;
        for (int e = tid; e < E; e += 256) {
            float v = fmaxf((float)p[e], 0.f);
            s0 += v; s1 += v * v;
        }
    }
    __shared__ float r0[256], r1[256];
    r0[tid] = s0; r1[tid] = s1; __syncthreads();
    for (int st = 128; st > 0; st >>= 1) {
        if (tid < st) { r0[tid] += r0[tid + st]; r1[tid] += r1[tid + st]; }
        __syncthreads();
    }
    if (tid == 0) {
        float N = (float)NB * (float)E;
        float m = r0[0] / N;
        float var = r1[0] / N - m * m;
        float sc = gamma[ch] * rsqrtf(var + 1e-5f);
        scsh[ch] = sc; scsh[C + ch] = beta[ch] - m * sc;
    }
}

// ---------------------------------------------------------------------------
// Mesh-conv GEMM with fused gather / BN-ReLU prologue and residual-ReLU
// epilogue.  D(Cout,E_tile) = Wk(Cout,K) x F(K,E_tile), f32 accum via
// v_wmma_f32_16x16x32_f16.  One workgroup: one batch, one 32-edge tile,
// ALL Cout (features gathered once).  8 waves; wave w owns edge sub-tile
// (w&1) and row tiles t*4 + (w>>1), t < NT = Cout/64.
// Template params make the K geometry and accumulator count compile-time:
//   KLOCAL: k-depth per channel chunk (32 for Cin==5, 160 otherwise)
//   CHUNKC: channels per chunk (5 or 32)
//   NT    : row tiles per wave (Cout/64: 1, 2 or 4)
// ---------------------------------------------------------------------------
template <int KLOCAL, int CHUNKC, int NT>
__global__ __launch_bounds__(256, 2)
void meshconv_wmma(const half_t* __restrict__ X, const half_t* __restrict__ Wk,
                   const int* __restrict__ gidx, const float* __restrict__ scsh,
                   const half_t* __restrict__ resid, half_t* __restrict__ Y,
                   int Cin, int Cout, int E) {
    constexpr int KSTEPS = KLOCAL / 32;
    constexpr int KSTR   = KLOCAL + 8;   // LDS row stride (168h = 84 dw: conflict-free)
    __shared__ __align__(16) half_t feat[TILE_E * KSTR];
    __shared__ int nbr[TILE_E * 4];

    const int tid  = threadIdx.x;
    const int lane = tid & 31;
    const int wave = tid >> 5;
    const int b     = blockIdx.z;
    const int eBase = blockIdx.x * TILE_E;

    const int nChunks = (CHUNKC == 5) ? 1 : (Cin >> 5);
    const int Kpad    = KLOCAL * nChunks;          // Wk row length

    // stage neighbor indices for this edge tile
    if (tid < TILE_E) {
        const int* g = gidx + ((size_t)b * E + (eBase + tid)) * 4;
        nbr[tid * 4 + 0] = g[0]; nbr[tid * 4 + 1] = g[1];
        nbr[tid * 4 + 2] = g[2]; nbr[tid * 4 + 3] = g[3];
    }
    if constexpr (CHUNKC == 5) {  // zero-pad K 25..32 (single chunk)
        for (int i = tid; i < TILE_E * KSTR; i += 256) feat[i] = (half_t)0.f;
    }
    __syncthreads();

    const int colSub = wave & 1;        // which 16-edge half this wave computes
    const int rowSub = wave >> 1;       // base row-tile (stride 4 across t)
    const int nCol   = lane & 15;       // N (edge) within tile; also M for A rows
    const int hi     = lane >> 4;
    const int mOff   = (lane < 16) ? 0 : 8;

    // hoisted fragment base pointers
    const half_t* bRow = feat + (colSub * 16 + nCol) * KSTR;   // B: [col][k]
    const half_t* aRow[NT];
#pragma unroll
    for (int t = 0; t < NT; ++t)
        aRow[t] = Wk + (size_t)((t * 4 + rowSub) * 16 + nCol) * Kpad + hi * 8;

    v8f acc[NT] = {};

    constexpr int ENTRIES = CHUNKC * TILE_E;
    for (int ch = 0; ch < nChunks; ++ch) {
        // ---- build gathered 5-feature columns (CHUNKC channels x 32 edges) ----
#pragma unroll
        for (int i0 = 0; i0 < ENTRIES; i0 += 256) {
            int i = i0 + tid;
            if (ENTRIES % 256 == 0 || i < ENTRIES) {
                int cl = i >> 5, e = i & 31;
                int c = ch * CHUNKC + cl;
                const half_t* xr = X + ((size_t)b * Cin + c) * E;
                float xv = (float)xr[eBase + e];
                float n1 = (float)xr[nbr[e * 4 + 0]];
                float n2 = (float)xr[nbr[e * 4 + 1]];
                float n3 = (float)xr[nbr[e * 4 + 2]];
                float n4 = (float)xr[nbr[e * 4 + 3]];
                if (scsh) {  // fused BN(relu(.)) prologue: v -> relu(v)*s + sh
                    float s = scsh[c], sh = scsh[Cin + c];
                    xv = fmaxf(xv, 0.f) * s + sh; n1 = fmaxf(n1, 0.f) * s + sh;
                    n2 = fmaxf(n2, 0.f) * s + sh; n3 = fmaxf(n3, 0.f) * s + sh;
                    n4 = fmaxf(n4, 0.f) * s + sh;
                }
                half_t* fr = feat + e * KSTR + cl;          // k = s*CHUNKC + cl
                fr[0 * CHUNKC] = (half_t)xv;
                fr[1 * CHUNKC] = (half_t)(n1 + n3);
                fr[2 * CHUNKC] = (half_t)(n2 + n4);
                fr[3 * CHUNKC] = (half_t)fabsf(n1 - n3);
                fr[4 * CHUNKC] = (half_t)fabsf(n2 - n4);
            }
        }
        __syncthreads();

        // ---- WMMA: KSTEPS x NT back-to-back matrix ops on the staged chunk ----
        const half_t* aChunkOff = (const half_t*)nullptr;  // (silence unused warn)
        (void)aChunkOff;
#pragma unroll
        for (int s = 0; s < KSTEPS; ++s) {
            // B fragment (32x16 f16): lane n<16 -> K {0..7,16..23},
            // lane n>=16 -> K {8..15,24..31} (relative to s*32)
            const half_t* q = bRow + s * 32 + hi * 8;
            v16h bf = concat8(*(const h8*)q, *(const h8*)(q + 16));
#pragma unroll
            for (int t = 0; t < NT; ++t) {
                // A fragment (16x32 f16): lane<16 -> K {0..7,16..23},
                // lane>=16 -> K {8..15,24..31}, row M = lane&15
                const half_t* p = aRow[t] + ch * KLOCAL + s * 32;
                v16h af = concat8(*(const h8*)p, *(const h8*)(p + 16));
                acc[t] = __builtin_amdgcn_wmma_f32_16x16x32_f16(
                    false, af, false, bf, (short)0, acc[t], false, false);
            }
        }
        __syncthreads();
    }

    // ---- epilogue: optional residual + relu, store f16 ----
    const int col = eBase + colSub * 16 + nCol;
    if (resid) {
#pragma unroll
        for (int t = 0; t < NT; ++t) {
            int rowBase = (t * 4 + rowSub) * 16;
#pragma unroll
            for (int r = 0; r < 8; ++r) {
                int row = rowBase + mOff + r;       // C/D layout: vgpr r, lane half
                size_t off = ((size_t)b * Cout + row) * E + col;
                Y[off] = (half_t)fmaxf(acc[t][r] + (float)resid[off], 0.f);
            }
        }
    } else {
#pragma unroll
        for (int t = 0; t < NT; ++t) {
            int rowBase = (t * 4 + rowSub) * 16;
#pragma unroll
            for (int r = 0; r < 8; ++r) {
                int row = rowBase + mOff + r;
                size_t off = ((size_t)b * Cout + row) * E + col;
                Y[off] = (half_t)acc[t][r];
            }
        }
    }
}

// ---------------------------------------------------------------------------
// MeshPool with fused outer-BN affine + relu:
//   Xn[b,c,e'] = relu( H[b,c,pool[b,e']] * s[c] + sh[c] )
// ---------------------------------------------------------------------------
__global__ void pool_bn(const half_t* __restrict__ H, const int* __restrict__ pool,
                        const float* __restrict__ scsh, half_t* __restrict__ Xn,
                        int C, int Eold, int Enew) {
    size_t i = (size_t)blockIdx.x * blockDim.x + threadIdx.x;
    size_t total = (size_t)NB * C * Enew;
    if (i >= total) return;
    int e = (int)(i % Enew);
    size_t t = i / Enew;
    int c = (int)(t % C), b = (int)(t / C);
    int src = pool[(size_t)b * Enew + e];
    float v = (float)H[((size_t)b * C + c) * Eold + src];
    Xn[i] = (half_t)fmaxf(v * scsh[c] + scsh[C + c], 0.f);
}

// ---------------------------------------------------------------------------
// Global average pool over edges: (16,256,2048) f16 -> (16,256) f32
// ---------------------------------------------------------------------------
__global__ void gap_kernel(const half_t* __restrict__ X, float* __restrict__ out) {
    int bc = blockIdx.x;                         // b*256 + c
    const half_t* p = X + (size_t)bc * 2048;
    float s = 0.f;
    for (int e = threadIdx.x; e < 2048; e += 256) s += (float)p[e];
    __shared__ float r[256];
    r[threadIdx.x] = s; __syncthreads();
    for (int st = 128; st > 0; st >>= 1) {
        if (threadIdx.x < st) r[threadIdx.x] += r[threadIdx.x + st];
        __syncthreads();
    }
    if (threadIdx.x == 0) out[bc] = r[0] * (1.f / 2048.f);
}

// ---------------------------------------------------------------------------
// Tiny dense layers (2 MFLOP total — not matrix-core material)
// ---------------------------------------------------------------------------
__global__ void fc_kernel(const float* __restrict__ in, const float* __restrict__ W,
                          const float* __restrict__ bias, float* __restrict__ out,
                          int In, int Out, int doRelu) {
    int i = blockIdx.x * blockDim.x + threadIdx.x;
    if (i >= NB * Out) return;
    int o = i % Out, b = i / Out;
    const float* w = W + (size_t)o * In;
    const float* x = in + (size_t)b * In;
    float s = bias[o];
    for (int k = 0; k < In; ++k) s += x[k] * w[k];
    out[i] = doRelu ? fmaxf(s, 0.f) : s;
}

// ---------------------------------------------------------------------------
extern "C" void kernel_launch(void* const* d_in, const int* in_sizes, int n_in,
                              void* d_out, int out_size, void* d_ws, size_t ws_size,
                              hipStream_t stream) {
    (void)n_in; (void)out_size; (void)ws_size;
    static const int Kc[5]   = {5, 64, 128, 256, 256};
    static const int RESc[5] = {16384, 12288, 8192, 4096, 2048};

    // --- input pointers (setup_inputs dict order) ---
    const float* x_in = (const float*)d_in[0];
    const int* gemmP[4]; const int* poolP[4];
    for (int i = 0; i < 4; ++i) { gemmP[i] = (const int*)d_in[1 + 2 * i];
                                  poolP[i] = (const int*)d_in[2 + 2 * i]; }
    // conv_ws: 16 arrays at [9..24] ; bn_ps: 24 at [25..48] ; norm_ps: 8 at [49..56]
    const float* fc1w = (const float*)d_in[57];
    const float* fc1b = (const float*)d_in[58];
    const float* fc2w = (const float*)d_in[59];
    const float* fc2b = (const float*)d_in[60];

    // --- workspace carve-out ---
    char* ws = (char*)d_ws;
    size_t off = 0;
    auto alloc = [&](size_t bytes) { size_t o = off; off = (off + bytes + 255) & ~(size_t)255; return ws + o; };
    const size_t MAXIN  = (size_t)NB * 256 * 4096;   // max block-input elems (f16)
    const size_t MAXOUT = (size_t)NB * 256 * 8192;   // max conv-output elems (f16)
    half_t* bufIn  = (half_t*)alloc(MAXIN  * sizeof(half_t));
    half_t* bufY0  = (half_t*)alloc(MAXOUT * sizeof(half_t));
    half_t* bufP0  = (half_t*)alloc(MAXOUT * sizeof(half_t));
    half_t* bufP1  = (half_t*)alloc(MAXOUT * sizeof(half_t));
    half_t* WkBuf  = (half_t*)alloc((size_t)256 * 1280 * sizeof(half_t));
    float*  scsh   = (float*)alloc(512 * sizeof(float));
    float*  gapBuf = (float*)alloc(NB * 256 * sizeof(float));
    float*  fc1out = (float*)alloc(NB * 512 * sizeof(float));

    // dispatch helper over the compile-time conv geometries
    auto launch_conv = [&](const half_t* X, const half_t* Wk, const int* gi,
                           const float* sc, const half_t* res, half_t* Yo,
                           int Cin, int Cout, int E) {
        dim3 grid(E / TILE_E, 1, NB);
        if (Cin == 5)           // Cout=64 -> NT=1
            meshconv_wmma<32, 5, 1><<<grid, 256, 0, stream>>>(X, Wk, gi, sc, res, Yo, Cin, Cout, E);
        else if (Cout == 64)
            meshconv_wmma<160, 32, 1><<<grid, 256, 0, stream>>>(X, Wk, gi, sc, res, Yo, Cin, Cout, E);
        else if (Cout == 128)
            meshconv_wmma<160, 32, 2><<<grid, 256, 0, stream>>>(X, Wk, gi, sc, res, Yo, Cin, Cout, E);
        else
            meshconv_wmma<160, 32, 4><<<grid, 256, 0, stream>>>(X, Wk, gi, sc, res, Yo, Cin, Cout, E);
    };

    // --- convert network input to f16 ---
    int n0 = in_sizes[0];
    cvt_f32_f16<<<(n0 + 255) / 256, 256, 0, stream>>>(x_in, bufIn, n0);

    // --- 4 MResConv blocks ---
    for (int i = 0; i < 4; ++i) {
        int Cin = Kc[i], Cout = Kc[i + 1], E = RESc[i], Enew = RESc[i + 1];

        // conv0 (no BN prologue, no residual)
        {
            int Kpad = (Cin == 5) ? 32 : 5 * Cin;
            const float* W0 = (const float*)d_in[9 + 4 * i + 0];
            prep_w<<<(Cout * Kpad + 255) / 256, 256, 0, stream>>>(W0, WkBuf, Cout, Cin, Kpad);
            launch_conv(bufIn, WkBuf, gemmP[i], nullptr, nullptr, bufY0, Cin, Cout, E);
        }
        half_t* cur = bufY0;
        for (int j = 1; j <= 3; ++j) {
            const float* g  = (const float*)d_in[25 + 6 * i + 2 * (j - 1)];
            const float* be = (const float*)d_in[25 + 6 * i + 2 * (j - 1) + 1];
            bn_stats<<<Cout, 256, 0, stream>>>(cur, g, be, scsh, Cout, E);
            int Kpad = 5 * Cout;
            const float* Wj = (const float*)d_in[9 + 4 * i + j];
            prep_w<<<(Cout * Kpad + 255) / 256, 256, 0, stream>>>(Wj, WkBuf, Cout, Cout, Kpad);
            half_t* outb = (j == 2) ? bufP1 : bufP0;
            const half_t* res = (j == 3) ? bufY0 : nullptr;   // fused relu(acc+x1)
            launch_conv(cur, WkBuf, gemmP[i], scsh, res, outb, Cout, Cout, E);
            cur = outb;
        }
        // outer BN (on h, already >=0 so relu-in-stats is a no-op) + relu + pool
        const float* ng = (const float*)d_in[49 + 2 * i];
        const float* nb = (const float*)d_in[49 + 2 * i + 1];
        bn_stats<<<Cout, 256, 0, stream>>>(cur, ng, nb, scsh, Cout, E);
        size_t tot = (size_t)NB * Cout * Enew;
        pool_bn<<<(int)((tot + 255) / 256), 256, 0, stream>>>(cur, poolP[i], scsh,
                                                              bufIn, Cout, E, Enew);
    }

    // --- head ---
    gap_kernel<<<NB * 256, 256, 0, stream>>>(bufIn, gapBuf);
    fc_kernel<<<(NB * 512 + 255) / 256, 256, 0, stream>>>(gapBuf, fc1w, fc1b, fc1out, 256, 512, 1);
    fc_kernel<<<(NB * 40 + 255) / 256, 256, 0, stream>>>(fc1out, fc2w, fc2b, (float*)d_out, 512, 40, 0);
}
